// GraphNetwork_89103391522964
// MI455X (gfx1250) — matrix operand order repaired and checked by reading, not compiled, not run on hardware
//
#include <hip/hip_runtime.h>
#include <hip/hip_bf16.h>
#include <stdint.h>

// ---------------------------------------------------------------------------
// MI455X (gfx1250) fused GraphNetwork block, v2.
//  * bf16 WMMA for all GEMMs (compute-bound: ~31.5 GFLOP vs ~50MB true HBM).
//  * Segment means via CSR gather (hist+scan+scatter, 0.64M int atomics)
//    instead of 123M scattered f32 atomics (which would dominate at ~200us).
//  * Edge kernel streams e1/e2 as bf16; e1 tile (contiguous 8KB) leaves LDS
//    through the CDNA5 async path: global_store_async_from_lds_b128.
//  * One WMMA tile per wave (no grid-stride loop) so the compiler cannot
//    hoist 64 weight fragments into >512 VGPRs like round 1 did.
// ---------------------------------------------------------------------------

typedef __attribute__((ext_vector_type(16))) __bf16 v16bf;
typedef __attribute__((ext_vector_type(8)))  __bf16 v8bf;
typedef __attribute__((ext_vector_type(8)))  float  v8f;

#define N_NODES 20000
#define N_EDGES 320000
#define N_GRAPHS 16
#define E_TILES (N_EDGES / 16)   // 20000
#define N_TILES (N_NODES / 16)   // 1250

// ---- WMMA helpers ----------------------------------------------------------

__device__ __forceinline__ v8f wmma_bf16(v16bf a, v16bf b, v8f c) {
  return __builtin_amdgcn_wmma_f32_16x16x32_bf16(
      false, a, false, b, (short)0, c, false, false);
}

// A fragment from row-major bf16 [M][K]; rowp -> row (m0 + lane%16), col k0.
// lanes<16 take K {0..15}, lanes>=16 take {16..31}. khalf=(lane>>4)*8
__device__ __forceinline__ v16bf loadA_bf16(const __bf16* __restrict__ rowp,
                                            int khalf) {
  v8bf lo = *(const v8bf*)(rowp + 2 * khalf);
  v8bf hi = *(const v8bf*)(rowp + 2 * khalf + 8);
  return __builtin_shufflevector(lo, hi, 0, 1, 2, 3, 4, 5, 6, 7,
                                 8, 9, 10, 11, 12, 13, 14, 15);
}

// B fragment: lane = column n (lane%16); lanes<16 hold K0..15, lanes>=16 K16..31.
__device__ __forceinline__ v16bf loadB_bf16(const __bf16* __restrict__ p) {
  v8bf lo = *(const v8bf*)(p);
  v8bf hi = *(const v8bf*)(p + 8);
  return __builtin_shufflevector(lo, hi, 0, 1, 2, 3, 4, 5, 6, 7,
                                 8, 9, 10, 11, 12, 13, 14, 15);
}

__device__ __forceinline__ v16bf cvtB_f32(const float* __restrict__ p, float s) {
  v16bf r;
#pragma unroll
  for (int i = 0; i < 16; ++i) r[i] = (__bf16)(p[i] * s);
  return r;
}

// ---- prep kernels ----------------------------------------------------------

__global__ void cvt_bf16_kernel(const float* __restrict__ src,
                                __bf16* __restrict__ dst, int n) {
  int i = blockIdx.x * blockDim.x + threadIdx.x;
  if (i < n) dst[i] = (__bf16)src[i];
}

__global__ void gemv16_kernel(const float* __restrict__ G,
                              const float* __restrict__ W,
                              float* __restrict__ out) {
  int i = blockIdx.x * blockDim.x + threadIdx.x;
  if (i >= N_GRAPHS * 128) return;
  int b = i >> 7, m = i & 127;
  float acc = 0.f;
#pragma unroll
  for (int k = 0; k < 16; ++k) acc += G[b * 16 + k] * W[m * 16 + k];
  out[i] = acc;
}

// ---- CSR build: histogram -> block scan -> scatter -------------------------

__global__ void hist_kernel(const int* __restrict__ receivers,
                            int* __restrict__ cnti) {
  int e = blockIdx.x * blockDim.x + threadIdx.x;
  if (e < N_EDGES) atomicAdd(&cnti[receivers[e]], 1);
}

__global__ __launch_bounds__(1024) void scan_kernel(const int* __restrict__ cnti,
                                                    int* __restrict__ woff,
                                                    int* __restrict__ wcur) {
  __shared__ int sums[1024];
  const int tid = threadIdx.x;
  const int per = (N_NODES + 1023) / 1024;  // 20
  const int base = tid * per;
  int s = 0;
  for (int i = 0; i < per; ++i) {
    int idx = base + i;
    if (idx < N_NODES) s += cnti[idx];
  }
  sums[tid] = s;
  __syncthreads();
  for (int d = 1; d < 1024; d <<= 1) {   // Hillis-Steele inclusive scan
    int v = (tid >= d) ? sums[tid - d] : 0;
    __syncthreads();
    sums[tid] += v;
    __syncthreads();
  }
  int run = (tid == 0) ? 0 : sums[tid - 1];
  for (int i = 0; i < per; ++i) {
    int idx = base + i;
    if (idx < N_NODES) {
      woff[idx] = run;
      wcur[idx] = run;
      run += cnti[idx];
    }
  }
  if (tid == 1023) woff[N_NODES] = run;  // == N_EDGES
}

__global__ void scatter_kernel(const int* __restrict__ receivers,
                               int* __restrict__ wcur,
                               int* __restrict__ elist) {
  int e = blockIdx.x * blockDim.x + threadIdx.x;
  if (e < N_EDGES) {
    int p = atomicAdd(&wcur[receivers[e]], 1);
    elist[p] = e;
  }
}

// ---- fused edge pipeline ----------------------------------------------------
// e1 = relu(E We1^T + be1) -> e1g (bf16, via async LDS->global DMA)
// e2 = relu(e1 We2^T + gWg2[g] + be2) -> e2g (bf16) + per-graph LDS sums.
__global__ __launch_bounds__(128) void edge_kernel(
    const float* __restrict__ edge_feats, const int* __restrict__ receivers,
    const int* __restrict__ node_graph, const __bf16* __restrict__ We1b,
    const float* __restrict__ be1, const __bf16* __restrict__ We2b,
    const float* __restrict__ be2, const float* __restrict__ gWg2,
    __bf16* __restrict__ e1g, __bf16* __restrict__ e2g,
    float* __restrict__ gesum, float* __restrict__ gecnt) {
  __shared__ __attribute__((aligned(16))) __bf16 e1buf[4][16 * 256];
  __shared__ float sSum[N_GRAPHS * 128];
  __shared__ float sCnt[N_GRAPHS];

  for (int i = threadIdx.x; i < N_GRAPHS * 128; i += blockDim.x) sSum[i] = 0.f;
  if (threadIdx.x < N_GRAPHS) sCnt[threadIdx.x] = 0.f;
  __syncthreads();

  const int lane  = threadIdx.x & 31;
  const int w     = threadIdx.x >> 5;
  const int col   = lane & 15;
  const int half  = lane >> 4;
  const int khalf = half * 8;
  const int t     = blockIdx.x * 4 + w;   // one 16-edge tile per wave

  if (t < E_TILES) {
    __bf16* myb = e1buf[w];
    const int edge = t * 16 + col;
    const int recv = receivers[edge];
    const int g    = node_graph[recv];
    v16bf b1 = cvtB_f32(edge_feats + edge * 32 + half * 16, 1.0f);

    // ---- GEMM1: e1^T, 16 out-tiles, K=32 in one WMMA each ----
#pragma unroll
    for (int j = 0; j < 16; ++j) {
      const int mbase = j * 16 + khalf;
      v16bf a1 = loadA_bf16(We1b + (j * 16 + col) * 32, khalf);
      v8f c;
#pragma unroll
      for (int r = 0; r < 8; ++r) c[r] = be1[mbase + r];
      c = wmma_bf16(a1, b1, c);
      v8bf p;
#pragma unroll
      for (int r = 0; r < 8; ++r) p[r] = (__bf16)fmaxf(c[r], 0.f);
      *(v8bf*)(myb + col * 256 + mbase) = p;   // ds_store_b128
    }

    // ---- push e1 tile (contiguous 8KB) to global via async LDS->global ----
    {
      asm volatile("s_wait_dscnt 0x0" ::: "memory");  // LDS writes done
      const unsigned lds_base = (unsigned)(uintptr_t)(void*)myb;
      const uint64_t gbase = (uint64_t)(uintptr_t)(e1g + (size_t)t * (16 * 256));
#pragma unroll
      for (int i = 0; i < 16; ++i) {
        unsigned la = lds_base + (unsigned)(lane * 16 + i * 512);
        uint64_t ga = gbase + (uint64_t)(lane * 16 + i * 512);
        asm volatile("global_store_async_from_lds_b128 %0, %1, off"
                     :: "v"(ga), "v"(la) : "memory");
      }
    }

    // B2 fragments of e1^T straight from LDS (lane reads its own edge row).
    v16bf b2f[8];
#pragma unroll
    for (int kk = 0; kk < 8; ++kk)
      b2f[kk] = loadB_bf16(myb + col * 256 + kk * 32 + half * 16);

    // ---- GEMM2: e2^T = We2 x e1^T (8 out-tiles x K=256) ----
#pragma unroll
    for (int j2 = 0; j2 < 8; ++j2) {
      const int mbase = j2 * 16 + khalf;
      v8f c2;
#pragma unroll
      for (int r = 0; r < 8; ++r)
        c2[r] = be2[mbase + r] + gWg2[g * 128 + mbase + r];
#pragma unroll
      for (int kk = 0; kk < 8; ++kk) {
        v16bf a2 = loadA_bf16(We2b + (j2 * 16 + col) * 256 + kk * 32, khalf);
        c2 = wmma_bf16(a2, b2f[kk], c2);
      }
      v8bf p2;
#pragma unroll
      for (int r = 0; r < 8; ++r) {
        float v = fmaxf(c2[r], 0.f);
        p2[r] = (__bf16)v;
        atomicAdd(&sSum[g * 128 + mbase + r], v);  // ds_add_f32
      }
      *(v8bf*)(e2g + (size_t)edge * 128 + mbase) = p2;
    }
    if (half == 0) atomicAdd(&sCnt[g], 1.0f);
  }

  __syncthreads();
  for (int i = threadIdx.x; i < N_GRAPHS * 128; i += blockDim.x)
    unsafeAtomicAdd(&gesum[i], sSum[i]);
  if (threadIdx.x < N_GRAPHS) unsafeAtomicAdd(&gecnt[threadIdx.x], sCnt[threadIdx.x]);
}

// ---- n1 = relu(N Wn1^T + mean(e1[in-edges]) Win1^T + bn1) -> n1ws bf16 -----
__global__ __launch_bounds__(256) void node1_kernel(
    const float* __restrict__ node_feats, const __bf16* __restrict__ e1g,
    const int* __restrict__ woff, const int* __restrict__ elist,
    const __bf16* __restrict__ Wn1b, const __bf16* __restrict__ Win1b,
    const float* __restrict__ bn1, __bf16* __restrict__ n1ws) {
  const int lane  = threadIdx.x & 31;
  const int col   = lane & 15;
  const int half  = lane >> 4;
  const int khalf = half * 8;
  const int t     = blockIdx.x * 8 + (threadIdx.x >> 5);
  if (t >= N_TILES) return;

  const int node = t * 16 + col;
  const int beg = woff[node], end = woff[node + 1];
  const float inv = 1.f / fmaxf((float)(end - beg), 1.f);

  v16bf bnf[2];
#pragma unroll
  for (int kk = 0; kk < 2; ++kk)
    bnf[kk] = cvtB_f32(node_feats + node * 64 + kk * 32 + half * 16, 1.f);

  // gather-mean of e1 rows into 8 B fragments, two 32-wide chunks per pass
  v16bf baf[8];
#pragma unroll
  for (int q = 0; q < 4; ++q) {
    float a0[16], a1[16];
#pragma unroll
    for (int r = 0; r < 16; ++r) { a0[r] = 0.f; a1[r] = 0.f; }
    for (int i = beg; i < end; ++i) {
      const __bf16* p = e1g + (size_t)elist[i] * 256 + q * 64 + half * 16;
      v8bf x0 = *(const v8bf*)(p);
      v8bf x1 = *(const v8bf*)(p + 8);
      v8bf y0 = *(const v8bf*)(p + 32);
      v8bf y1 = *(const v8bf*)(p + 40);
#pragma unroll
      for (int r = 0; r < 8; ++r) {
        a0[r] += (float)x0[r]; a0[8 + r] += (float)x1[r];
        a1[r] += (float)y0[r]; a1[8 + r] += (float)y1[r];
      }
    }
    v16bf f0, f1;
#pragma unroll
    for (int r = 0; r < 16; ++r) {
      f0[r] = (__bf16)(a0[r] * inv);
      f1[r] = (__bf16)(a1[r] * inv);
    }
    baf[2 * q] = f0;
    baf[2 * q + 1] = f1;
  }

#pragma unroll
  for (int j = 0; j < 16; ++j) {
    const int mbase = j * 16 + khalf;
    v8f c;
#pragma unroll
    for (int r = 0; r < 8; ++r) c[r] = bn1[mbase + r];
#pragma unroll
    for (int kk = 0; kk < 2; ++kk)
      c = wmma_bf16(loadA_bf16(Wn1b + (j * 16 + col) * 64 + kk * 32, khalf),
                    bnf[kk], c);
#pragma unroll
    for (int kk = 0; kk < 8; ++kk)
      c = wmma_bf16(loadA_bf16(Win1b + (j * 16 + col) * 256 + kk * 32, khalf),
                    baf[kk], c);
    v8bf p;
#pragma unroll
    for (int r = 0; r < 8; ++r) p[r] = (__bf16)fmaxf(c[r], 0.f);
    *(v8bf*)(n1ws + (size_t)node * 256 + mbase) = p;
  }
}

// ---- n2 = relu(n1 Wn2^T + mean(e2[in]) Win2^T + gWng2[g] + bn2); graph sums
__global__ __launch_bounds__(256) void node2_kernel(
    const __bf16* __restrict__ n1ws, const __bf16* __restrict__ e2g,
    const int* __restrict__ woff, const int* __restrict__ elist,
    const int* __restrict__ node_graph, const __bf16* __restrict__ Wn2b,
    const __bf16* __restrict__ Win2b, const float* __restrict__ bn2,
    const float* __restrict__ gWng2, float* __restrict__ gnsum,
    float* __restrict__ gncnt) {
  __shared__ float sSum[N_GRAPHS * 128];
  __shared__ float sCnt[N_GRAPHS];
  for (int i = threadIdx.x; i < N_GRAPHS * 128; i += blockDim.x) sSum[i] = 0.f;
  if (threadIdx.x < N_GRAPHS) sCnt[threadIdx.x] = 0.f;
  __syncthreads();

  const int lane  = threadIdx.x & 31;
  const int col   = lane & 15;
  const int half  = lane >> 4;
  const int khalf = half * 8;
  const int t     = blockIdx.x * 8 + (threadIdx.x >> 5);

  if (t < N_TILES) {
    const int node = t * 16 + col;
    const int g    = node_graph[node];
    const int beg = woff[node], end = woff[node + 1];
    const float inv = 1.f / fmaxf((float)(end - beg), 1.f);

    v16bf b1f[8];
#pragma unroll
    for (int kk = 0; kk < 8; ++kk)
      b1f[kk] = loadB_bf16(n1ws + (size_t)node * 256 + kk * 32 + half * 16);

    v16bf b2f[4];
#pragma unroll
    for (int q = 0; q < 2; ++q) {
      float a0[16], a1[16];
#pragma unroll
      for (int r = 0; r < 16; ++r) { a0[r] = 0.f; a1[r] = 0.f; }
      for (int i = beg; i < end; ++i) {
        const __bf16* p = e2g + (size_t)elist[i] * 128 + q * 64 + half * 16;
        v8bf x0 = *(const v8bf*)(p);
        v8bf x1 = *(const v8bf*)(p + 8);
        v8bf y0 = *(const v8bf*)(p + 32);
        v8bf y1 = *(const v8bf*)(p + 40);
#pragma unroll
        for (int r = 0; r < 8; ++r) {
          a0[r] += (float)x0[r]; a0[8 + r] += (float)x1[r];
          a1[r] += (float)y0[r]; a1[8 + r] += (float)y1[r];
        }
      }
      v16bf f0, f1;
#pragma unroll
      for (int r = 0; r < 16; ++r) {
        f0[r] = (__bf16)(a0[r] * inv);
        f1[r] = (__bf16)(a1[r] * inv);
      }
      b2f[2 * q] = f0;
      b2f[2 * q + 1] = f1;
    }

#pragma unroll
    for (int j = 0; j < 8; ++j) {
      const int mbase = j * 16 + khalf;
      v8f c;
#pragma unroll
      for (int r = 0; r < 8; ++r)
        c[r] = bn2[mbase + r] + gWng2[g * 128 + mbase + r];
#pragma unroll
      for (int kk = 0; kk < 8; ++kk)
        c = wmma_bf16(loadA_bf16(Wn2b + (j * 16 + col) * 256 + kk * 32, khalf),
                      b1f[kk], c);
#pragma unroll
      for (int kk = 0; kk < 4; ++kk)
        c = wmma_bf16(loadA_bf16(Win2b + (j * 16 + col) * 128 + kk * 32, khalf),
                      b2f[kk], c);
#pragma unroll
      for (int r = 0; r < 8; ++r)
        atomicAdd(&sSum[g * 128 + mbase + r], fmaxf(c[r], 0.f));
    }
    if (half == 0) atomicAdd(&sCnt[g], 1.0f);
  }
  __syncthreads();
  for (int i = threadIdx.x; i < N_GRAPHS * 128; i += blockDim.x)
    unsafeAtomicAdd(&gnsum[i], sSum[i]);
  if (threadIdx.x < N_GRAPHS) unsafeAtomicAdd(&gncnt[threadIdx.x], sCnt[threadIdx.x]);
}

// ---- final combine (tiny 16x128) -------------------------------------------
__global__ void final_kernel(const float* __restrict__ gnsum,
                             const float* __restrict__ gncnt,
                             const float* __restrict__ gesum,
                             const float* __restrict__ gecnt,
                             const float* __restrict__ G,
                             const float* __restrict__ Wgn,
                             const float* __restrict__ Wge,
                             const float* __restrict__ Wgg,
                             const float* __restrict__ bg,
                             float* __restrict__ out) {
  int i = blockIdx.x * blockDim.x + threadIdx.x;
  if (i >= N_GRAPHS * 128) return;
  int b = i >> 7, m = i & 127;
  float invn = 1.f / fmaxf(gncnt[b], 1.f);
  float inve = 1.f / fmaxf(gecnt[b], 1.f);
  float acc = bg[m];
  for (int k = 0; k < 128; ++k)
    acc += gnsum[b * 128 + k] * invn * Wgn[m * 128 + k] +
           gesum[b * 128 + k] * inve * Wge[m * 128 + k];
#pragma unroll
  for (int k = 0; k < 16; ++k) acc += G[b * 16 + k] * Wgg[m * 16 + k];
  out[i] = acc;
}

// ---------------------------------------------------------------------------

extern "C" void kernel_launch(void* const* d_in, const int* in_sizes, int n_in,
                              void* d_out, int out_size, void* d_ws, size_t ws_size,
                              hipStream_t stream) {
  (void)in_sizes; (void)n_in; (void)out_size; (void)ws_size;
  const float* node_feats = (const float*)d_in[0];
  const float* edge_feats = (const float*)d_in[1];
  const float* globals_   = (const float*)d_in[2];
  const float* We1  = (const float*)d_in[3];
  const float* be1  = (const float*)d_in[4];
  const float* Wn1  = (const float*)d_in[5];
  const float* Win1 = (const float*)d_in[6];
  const float* bn1  = (const float*)d_in[7];
  const float* We2  = (const float*)d_in[8];
  const float* Wg2  = (const float*)d_in[9];
  const float* be2  = (const float*)d_in[10];
  const float* Wn2  = (const float*)d_in[11];
  const float* Win2 = (const float*)d_in[12];
  const float* Wng2 = (const float*)d_in[13];
  const float* bn2  = (const float*)d_in[14];
  const float* Wgn  = (const float*)d_in[15];
  const float* Wge  = (const float*)d_in[16];
  const float* Wgg  = (const float*)d_in[17];
  const float* bg   = (const float*)d_in[18];
  const int* receivers  = (const int*)d_in[19];
  const int* node_graph = (const int*)d_in[20];
  float* out = (float*)d_out;

  // ---- workspace layout (16B-aligned slabs) ----
  char* w = (char*)d_ws;
  size_t off = 0;
  int*   cnti  = (int*)(w + off);   off += (size_t)N_NODES * 4;       // zeroed
  float* gesum = (float*)(w + off); off += N_GRAPHS * 128 * 4;        // zeroed
  float* gnsum = (float*)(w + off); off += N_GRAPHS * 128 * 4;        // zeroed
  float* gecnt = (float*)(w + off); off += N_GRAPHS * 4;              // zeroed
  float* gncnt = (float*)(w + off); off += N_GRAPHS * 4;              // zeroed
  size_t zero_bytes = off;
  int*   woff  = (int*)(w + off);   off += 80016;  // N_NODES+1 ints, padded
  int*   wcur  = (int*)(w + off);   off += (size_t)N_NODES * 4;
  int*   elist = (int*)(w + off);   off += (size_t)N_EDGES * 4;
  __bf16* e1g  = (__bf16*)(w + off); off += (size_t)N_EDGES * 256 * 2; // 164MB
  __bf16* e2g  = (__bf16*)(w + off); off += (size_t)N_EDGES * 128 * 2; // 82MB
  __bf16* n1ws = (__bf16*)(w + off); off += (size_t)N_NODES * 256 * 2;
  __bf16* We1b  = (__bf16*)(w + off); off += 256 * 32 * 2;
  __bf16* Wn1b  = (__bf16*)(w + off); off += 256 * 64 * 2;
  __bf16* Win1b = (__bf16*)(w + off); off += 256 * 256 * 2;
  __bf16* We2b  = (__bf16*)(w + off); off += 128 * 256 * 2;
  __bf16* Wn2b  = (__bf16*)(w + off); off += 128 * 256 * 2;
  __bf16* Win2b = (__bf16*)(w + off); off += 128 * 128 * 2;
  float* gWg2  = (float*)(w + off); off += N_GRAPHS * 128 * 4;
  float* gWng2 = (float*)(w + off); off += N_GRAPHS * 128 * 4;

  hipMemsetAsync(d_ws, 0, zero_bytes, stream);

  auto cvt = [&](const float* s, __bf16* d, int n) {
    cvt_bf16_kernel<<<(n + 255) / 256, 256, 0, stream>>>(s, d, n);
  };
  cvt(We1, We1b, 256 * 32);
  cvt(Wn1, Wn1b, 256 * 64);
  cvt(Win1, Win1b, 256 * 256);
  cvt(We2, We2b, 128 * 256);
  cvt(Wn2, Wn2b, 128 * 256);
  cvt(Win2, Win2b, 128 * 128);

  gemv16_kernel<<<8, 256, 0, stream>>>(globals_, Wg2, gWg2);
  gemv16_kernel<<<8, 256, 0, stream>>>(globals_, Wng2, gWng2);

  // CSR build
  hist_kernel<<<(N_EDGES + 255) / 256, 256, 0, stream>>>(receivers, cnti);
  scan_kernel<<<1, 1024, 0, stream>>>(cnti, woff, wcur);
  scatter_kernel<<<(N_EDGES + 255) / 256, 256, 0, stream>>>(receivers, wcur, elist);

  // fused edge pipeline: one 16-edge tile per wave
  edge_kernel<<<E_TILES / 4, 128, 0, stream>>>(edge_feats, receivers, node_graph,
                                               We1b, be1, We2b, be2, gWg2,
                                               e1g, e2g, gesum, gecnt);

  node1_kernel<<<(N_TILES + 7) / 8, 256, 0, stream>>>(
      node_feats, e1g, woff, elist, Wn1b, Win1b, bn1, n1ws);

  node2_kernel<<<(N_TILES + 7) / 8, 256, 0, stream>>>(
      n1ws, e2g, woff, elist, node_graph, Wn2b, Win2b, bn2, gWng2, gnsum, gncnt);

  final_kernel<<<8, 256, 0, stream>>>(gnsum, gncnt, gesum, gecnt, globals_,
                                      Wgn, Wge, Wgg, bg, out);
}
